// KimiSparseMoeBlock_68195490726076
// MI455X (gfx1250) — matrix operand order repaired and checked by reading, not compiled, not run on hardware
//
#include <hip/hip_runtime.h>
#include <math.h>

typedef __attribute__((ext_vector_type(16))) __bf16 v16bf;
typedef __attribute__((ext_vector_type(8)))  __bf16 v8bf;
typedef __attribute__((ext_vector_type(4)))  __bf16 v4bf;
typedef __attribute__((ext_vector_type(8)))  float  v8f;

#define E_    32
#define K_    4
#define G_    4
#define TG_   2
#define H_    2048
#define F_    1024
#define FS_   2048
#define NTOK  2048
#define CAP_  512
#define SCALE_ 2.0f
#define LP    40      // LDS pitch (bf16 elems): 80B rows, 16B-aligned, conflict-skewed

// ---- async global->LDS copy (CDNA5 path), guarded for toolchain support ----
#if __has_builtin(__builtin_amdgcn_global_load_async_to_lds_b128)
#define HAVE_ASYNC_LDS 1
typedef int v4i_t __attribute__((vector_size(16)));
static __device__ __forceinline__ void cp_async_b128(const __bf16* g, __bf16* l) {
  __builtin_amdgcn_global_load_async_to_lds_b128((v4i_t*)g, (v4i_t*)l, 0, 0);
}
#endif
static __device__ __forceinline__ void wait_async_all() {
#if HAVE_ASYNC_LDS
#if __has_builtin(__builtin_amdgcn_s_wait_asynccnt)
  __builtin_amdgcn_s_wait_asynccnt(0);
#else
  asm volatile("s_wait_asynccnt 0x0" ::: "memory");
#endif
#endif
}

static __device__ __forceinline__ __bf16 f2bf(float f) {
  unsigned u = __builtin_bit_cast(unsigned, f);
  unsigned short s = (unsigned short)((u + 0x8000u) >> 16);   // round-half-up
  return __builtin_bit_cast(__bf16, s);
}

// pack two f32 into (lo16 = bf16(a), hi16 = bf16(b))
static __device__ __forceinline__ unsigned bfpack2(float a, float b) {
  unsigned ua = __builtin_bit_cast(unsigned, a);
  unsigned ub = __builtin_bit_cast(unsigned, b);
  return ((ua + 0x8000u) >> 16) | ((ub + 0x8000u) & 0xffff0000u);
}

// ---------------- f32 -> bf16 elementwise ----------------
__global__ void k_cvt_bf16(const float* __restrict__ x, __bf16* __restrict__ y) {
  size_t i = ((size_t)blockIdx.x * 256 + threadIdx.x) * 4;
  float4 v = *(const float4*)(x + i);
  v4bf o; o[0] = f2bf(v.x); o[1] = f2bf(v.y); o[2] = f2bf(v.z); o[3] = f2bf(v.w);
  *(v4bf*)(y + i) = o;
}

// ---------------- gate: sigmoid + grouped top-k ----------------
__global__ __launch_bounds__(128) void k_gate(const float* __restrict__ x,
                                              const float* __restrict__ gw,
                                              const float* __restrict__ gb,
                                              int* __restrict__ tidx,
                                              float* __restrict__ tw) {
  __shared__ float xs[4][H_];
  __shared__ float s_score[4][E_];
  __shared__ float s_sc[4][E_];
  int tid = threadIdx.x, wv = tid >> 5, lane = tid & 31;
  int n = blockIdx.x * 4 + wv;

  const float4* xr = (const float4*)(x + (size_t)n * H_);
  float4* xd = (float4*)&xs[wv][0];
  for (int i = lane; i < H_ / 4; i += 32) xd[i] = xr[i];
  __syncthreads();

  const float4* w4 = (const float4*)(gw + (size_t)lane * H_);
  float acc = 0.f;
  for (int i = 0; i < H_ / 4; i++) {
    float4 a = xd[i]; float4 b = w4[i];
    acc += a.x * b.x + a.y * b.y + a.z * b.z + a.w * b.w;
  }
  float sc = 1.f / (1.f + __expf(-acc));
  s_score[wv][lane] = sc;
  s_sc[wv][lane] = sc + gb[lane];
  __syncthreads();

  if (lane == 0) {
    float gs[G_];
    for (int g = 0; g < G_; g++) {
      float m1 = -1e30f, m2 = -1e30f;
      for (int j = 0; j < E_ / G_; j++) {
        float v = s_sc[wv][g * (E_ / G_) + j];
        if (v > m1) { m2 = m1; m1 = v; } else if (v > m2) { m2 = v; }
      }
      gs[g] = m1 + m2;
    }
    bool gsel[G_] = {};
    for (int t = 0; t < TG_; t++) {
      int bg = 0; float best = -1e30f;
      for (int g = 0; g < G_; g++)
        if (!gsel[g] && gs[g] > best) { best = gs[g]; bg = g; }
      gsel[bg] = true;
    }
    bool taken[E_] = {};
    int ids[K_]; float wk[K_]; float wsum = 0.f;
    for (int k = 0; k < K_; k++) {
      int bi = 0; float best = -1e30f;
      for (int e = 0; e < E_; e++) {
        if (taken[e]) continue;
        float v = gsel[e / (E_ / G_)] ? s_sc[wv][e] : 0.0f;   // masked -> 0.0 (matches ref)
        if (v > best) { best = v; bi = e; }
      }
      taken[bi] = true;
      ids[k] = bi; wk[k] = s_score[wv][bi]; wsum += wk[k];
    }
    float inv = SCALE_ / (wsum + 1e-20f);
    for (int k = 0; k < K_; k++) {
      tidx[n * K_ + k] = ids[k];
      tw[n * K_ + k]   = wk[k] * inv;
    }
  }
}

// ---------------- dispatch ----------------
__global__ void k_reset(int* __restrict__ cnt) {
  if (threadIdx.x < E_) cnt[threadIdx.x] = 0;
}

__global__ __launch_bounds__(256) void k_scatter(const float* __restrict__ x,
                                                 const int* __restrict__ tidx,
                                                 int* __restrict__ cnt,
                                                 __bf16* __restrict__ bufx,
                                                 int* __restrict__ slots) {
  __shared__ int sh_e[8], sh_s[8];
  int tid = threadIdx.x, wv = tid >> 5, lane = tid & 31;
  int p = blockIdx.x * 8 + wv;
  if (lane == 0) {
    int e = tidx[p];
    int s = atomicAdd(&cnt[e], 1);
    sh_e[wv] = e; sh_s[wv] = s;
    slots[p] = (s < CAP_) ? s : -1;
  }
  __syncthreads();
  int e = sh_e[wv], s = sh_s[wv];
  if (s >= CAP_) return;
  int n = p >> 2;                                     // p / K_
  const float4* src = (const float4*)(x + (size_t)n * H_);
  __bf16* dst = bufx + ((size_t)e * CAP_ + s) * H_;
  for (int it = 0; it < 16; it++) {
    int i = it * 32 + lane;
    float4 v = src[i];
    v4bf o; o[0] = f2bf(v.x); o[1] = f2bf(v.y); o[2] = f2bf(v.z); o[3] = f2bf(v.w);
    *(v4bf*)(dst + (size_t)i * 4) = o;
  }
}

// ==================== WMMA GEMMs ====================
// Block tile 128(M) x 64(N), k-step 32, 8 waves each owning a 32x32 output
// tile (2x2 v_wmma_f32_16x16x32_bf16). LDS double-buffered, one barrier/step.
// A: bf16 [M x Kd], copied global->LDS (async path when available).
// W: f32 [Kd x Nd], staged as bf16 k-major rows via column-segment loads.

union Frag { v16bf v; v8bf h[2]; };

#if HAVE_ASYNC_LDS
#define LOAD_A(K0, BUF)                                                       \
  { const __bf16* p = ap + (K0);                                              \
    __bf16* l = &sA[BUF][ar * LP + ac];                                       \
    cp_async_b128(p, l); cp_async_b128(p + 8, l + 8); }
#define STORE_A(BUF)
#else
#define LOAD_A(K0, BUF)                                                       \
  { const __bf16* p = ap + (K0);                                              \
    ra0 = *(const uint4*)p; ra1 = *(const uint4*)(p + 8); }
#define STORE_A(BUF)                                                          \
  { *(uint4*)&sA[BUF][ar * LP + ac] = ra0;                                    \
    *(uint4*)&sA[BUF][ar * LP + ac + 8] = ra1; }
#endif

__global__ __launch_bounds__(256) void k_gemm_swiglu(
    const __bf16* __restrict__ A, const float* __restrict__ W1,
    const float* __restrict__ W3, __bf16* __restrict__ O,
    int Kd, int Nd, long sAe, long sWe, long sOe) {
  __shared__ __bf16 sA[2][128 * LP];
  __shared__ __bf16 sB1[2][64 * LP];
  __shared__ __bf16 sB3[2][64 * LP];
  int tid = threadIdx.x;
  int n0 = blockIdx.x * 64, m0 = blockIdx.y * 128;
  long ez = blockIdx.z;
  const __bf16* Ae = A + ez * sAe;
  const float* W1e = W1 + ez * sWe;
  const float* W3e = W3 + ez * sWe;
  __bf16* Oe = O + ez * sOe;

  int wid = tid >> 5, lane = tid & 31;
  int wm = wid & 3, wn = wid >> 2;          // wave tile: rows wm*32, cols wn*32
  int half = lane >> 4, mr = lane & 15;

  int ar = tid >> 1, ac = (tid & 1) * 16;   // A staging: 128 rows x 32 cols
  int bn = tid & 63, bkc = (tid >> 6) * 8;  // W staging: col n, k-run of 8

  v8f acc1[2][2] = {}; v8f acc3[2][2] = {};
  uint4 ra0 = {}, ra1 = {};                  // A stage regs (fallback path)
  float v1r[8], v3r[8];                      // W stage regs

  const __bf16* ap = Ae + (size_t)(m0 + ar) * Kd + ac;
  const float* w1p = W1e + (size_t)bkc * Nd + n0 + bn;
  const float* w3p = W3e + (size_t)bkc * Nd + n0 + bn;

#define LOAD_W(K0)                                                            \
  { _Pragma("unroll") for (int j = 0; j < 8; j++) {                           \
      v1r[j] = w1p[(size_t)((K0) + j) * Nd];                                  \
      v3r[j] = w3p[(size_t)((K0) + j) * Nd];                                  \
    } }

#define STORE_W(BUF)                                                          \
  { uint4 p1, p3;                                                             \
    p1.x = bfpack2(v1r[0], v1r[1]); p1.y = bfpack2(v1r[2], v1r[3]);           \
    p1.z = bfpack2(v1r[4], v1r[5]); p1.w = bfpack2(v1r[6], v1r[7]);           \
    p3.x = bfpack2(v3r[0], v3r[1]); p3.y = bfpack2(v3r[2], v3r[3]);           \
    p3.z = bfpack2(v3r[4], v3r[5]); p3.w = bfpack2(v3r[6], v3r[7]);           \
    *(uint4*)&sB1[BUF][bn * LP + bkc] = p1;                                   \
    *(uint4*)&sB3[BUF][bn * LP + bkc] = p3; }

  LOAD_A(0, 0); LOAD_W(0);
  STORE_A(0);   STORE_W(0);
  wait_async_all();
  __syncthreads();

  int nk = Kd >> 5;
  for (int t = 0; t < nk; t++) {
    int cur = t & 1;
    bool more = (t + 1) < nk;
    if (more) { LOAD_A((t + 1) << 5, cur ^ 1); LOAD_W((t + 1) << 5); }

    Frag af[2];
#pragma unroll
    for (int mt = 0; mt < 2; mt++) {
      int ab = (wm * 32 + mt * 16 + mr) * LP + half * 8;  // A: K-halves interleaved
      af[mt].h[0] = *(const v8bf*)&sA[cur][ab];
      af[mt].h[1] = *(const v8bf*)&sA[cur][ab + 16];
    }
#pragma unroll
    for (int nt = 0; nt < 2; nt++) {
      int bb = (wn * 32 + nt * 16 + mr) * LP + half * 8;  // B: K-halves interleaved
      Frag b1, b3;
      b1.h[0] = *(const v8bf*)&sB1[cur][bb]; b1.h[1] = *(const v8bf*)&sB1[cur][bb + 16];
      b3.h[0] = *(const v8bf*)&sB3[cur][bb]; b3.h[1] = *(const v8bf*)&sB3[cur][bb + 16];
#pragma unroll
      for (int mt = 0; mt < 2; mt++) {
        acc1[mt][nt] = __builtin_amdgcn_wmma_f32_16x16x32_bf16(
            false, af[mt].v, false, b1.v, (short)0, acc1[mt][nt], false, false);
        acc3[mt][nt] = __builtin_amdgcn_wmma_f32_16x16x32_bf16(
            false, af[mt].v, false, b3.v, (short)0, acc3[mt][nt], false, false);
      }
    }
    if (more) { STORE_A(cur ^ 1); STORE_W(cur ^ 1); }
    wait_async_all();
    __syncthreads();
  }

#pragma unroll
  for (int mt = 0; mt < 2; mt++)
#pragma unroll
    for (int nt = 0; nt < 2; nt++) {
      int ng = n0 + wn * 32 + nt * 16 + mr;
#pragma unroll
      for (int r = 0; r < 8; r++) {
        int mg = m0 + wm * 32 + mt * 16 + half * 8 + r;
        float c1 = acc1[mt][nt][r], c3 = acc3[mt][nt][r];
        float hv = (c1 / (1.f + __expf(-c1))) * c3;     // silu(c1) * c3
        Oe[(size_t)mg * Nd + ng] = f2bf(hv);
      }
    }
#undef LOAD_W
#undef STORE_W
}

__global__ __launch_bounds__(256) void k_gemm_plain(
    const __bf16* __restrict__ A, const float* __restrict__ W,
    float* __restrict__ O, int Kd, int Nd, long sAe, long sWe, long sOe) {
  __shared__ __bf16 sA[2][128 * LP];
  __shared__ __bf16 sB[2][64 * LP];
  int tid = threadIdx.x;
  int n0 = blockIdx.x * 64, m0 = blockIdx.y * 128;
  long ez = blockIdx.z;
  const __bf16* Ae = A + ez * sAe;
  const float* We = W + ez * sWe;
  float* Oe = O + ez * sOe;

  int wid = tid >> 5, lane = tid & 31;
  int wm = wid & 3, wn = wid >> 2;
  int half = lane >> 4, mr = lane & 15;
  int ar = tid >> 1, ac = (tid & 1) * 16;
  int bn = tid & 63, bkc = (tid >> 6) * 8;

  v8f acc[2][2] = {};
  uint4 ra0 = {}, ra1 = {};
  float vr[8];

  const __bf16* ap = Ae + (size_t)(m0 + ar) * Kd + ac;
  const float* wp = We + (size_t)bkc * Nd + n0 + bn;

#define LOAD_W(K0)                                                            \
  { _Pragma("unroll") for (int j = 0; j < 8; j++)                             \
      vr[j] = wp[(size_t)((K0) + j) * Nd]; }

#define STORE_W(BUF)                                                          \
  { uint4 pk;                                                                 \
    pk.x = bfpack2(vr[0], vr[1]); pk.y = bfpack2(vr[2], vr[3]);               \
    pk.z = bfpack2(vr[4], vr[5]); pk.w = bfpack2(vr[6], vr[7]);               \
    *(uint4*)&sB[BUF][bn * LP + bkc] = pk; }

  LOAD_A(0, 0); LOAD_W(0);
  STORE_A(0);   STORE_W(0);
  wait_async_all();
  __syncthreads();

  int nk = Kd >> 5;
  for (int t = 0; t < nk; t++) {
    int cur = t & 1;
    bool more = (t + 1) < nk;
    if (more) { LOAD_A((t + 1) << 5, cur ^ 1); LOAD_W((t + 1) << 5); }

    Frag af[2];
#pragma unroll
    for (int mt = 0; mt < 2; mt++) {
      int ab = (wm * 32 + mt * 16 + mr) * LP + half * 8;
      af[mt].h[0] = *(const v8bf*)&sA[cur][ab];
      af[mt].h[1] = *(const v8bf*)&sA[cur][ab + 16];
    }
#pragma unroll
    for (int nt = 0; nt < 2; nt++) {
      int bb = (wn * 32 + nt * 16 + mr) * LP + half * 8;
      Frag fb;
      fb.h[0] = *(const v8bf*)&sB[cur][bb]; fb.h[1] = *(const v8bf*)&sB[cur][bb + 16];
#pragma unroll
      for (int mt = 0; mt < 2; mt++)
        acc[mt][nt] = __builtin_amdgcn_wmma_f32_16x16x32_bf16(
            false, af[mt].v, false, fb.v, (short)0, acc[mt][nt], false, false);
    }
    if (more) { STORE_A(cur ^ 1); STORE_W(cur ^ 1); }
    wait_async_all();
    __syncthreads();
  }

#pragma unroll
  for (int mt = 0; mt < 2; mt++)
#pragma unroll
    for (int nt = 0; nt < 2; nt++) {
      int ng = n0 + wn * 32 + nt * 16 + mr;
#pragma unroll
      for (int r = 0; r < 8; r++) {
        int mg = m0 + wm * 32 + mt * 16 + half * 8 + r;
        Oe[(size_t)mg * Nd + ng] = acc[mt][nt][r];
      }
    }
#undef LOAD_W
#undef STORE_W
}

// ---------------- combine: shared + sum_k w_k * expert_out ----------------
__global__ __launch_bounds__(256) void k_combine(const float* __restrict__ shout,
                                                 const float* __restrict__ obuf,
                                                 const int* __restrict__ tidx,
                                                 const float* __restrict__ tw,
                                                 const int* __restrict__ slots,
                                                 float* __restrict__ out) {
  int n = blockIdx.x;
  int h0 = threadIdx.x * 8;
  size_t base = (size_t)n * H_ + h0;
  float4 a0 = *(const float4*)(shout + base);
  float4 a1 = *(const float4*)(shout + base + 4);
  for (int k = 0; k < K_; k++) {
    int p = n * K_ + k;
    int s = slots[p];
    if (s < 0) continue;
    int e = tidx[p];
    float w = tw[p];
    const float* row = obuf + ((size_t)e * CAP_ + s) * H_ + h0;
    float4 b0 = *(const float4*)row;
    float4 b1 = *(const float4*)(row + 4);
    a0.x += w * b0.x; a0.y += w * b0.y; a0.z += w * b0.z; a0.w += w * b0.w;
    a1.x += w * b1.x; a1.y += w * b1.y; a1.z += w * b1.z; a1.w += w * b1.w;
  }
  *(float4*)(out + base) = a0;
  *(float4*)(out + base + 4) = a1;
}

static inline size_t align256(size_t x) { return (x + 255) & ~(size_t)255; }

extern "C" void kernel_launch(void* const* d_in, const int* in_sizes, int n_in,
                              void* d_out, int out_size, void* d_ws, size_t ws_size,
                              hipStream_t stream) {
  (void)in_sizes; (void)n_in; (void)out_size; (void)ws_size;
  const float* x          = (const float*)d_in[0];   // [B,S,H]
  const float* gate_w     = (const float*)d_in[1];   // [E,H]
  const float* gate_bias  = (const float*)d_in[2];   // [E]
  const float* w1         = (const float*)d_in[3];   // [E,H,F]
  const float* w2         = (const float*)d_in[4];   // [E,F,H]
  const float* w3         = (const float*)d_in[5];   // [E,H,F]
  const float* sh_gate    = (const float*)d_in[6];   // [H,FS]
  const float* sh_up      = (const float*)d_in[7];   // [H,FS]
  const float* sh_down    = (const float*)d_in[8];   // [FS,H]
  float* out = (float*)d_out;

  char* ws = (char*)d_ws;
  size_t o = 0;
  __bf16* xbf   = (__bf16*)(ws + o); o = align256(o + (size_t)NTOK * H_ * 2);
  int*    tidx  = (int*)   (ws + o); o = align256(o + (size_t)NTOK * K_ * 4);
  float*  tw    = (float*) (ws + o); o = align256(o + (size_t)NTOK * K_ * 4);
  int*    slots = (int*)   (ws + o); o = align256(o + (size_t)NTOK * K_ * 4);
  int*    cnt   = (int*)   (ws + o); o = align256(o + (size_t)E_ * 4);
  __bf16* bufx  = (__bf16*)(ws + o); o = align256(o + (size_t)E_ * CAP_ * H_ * 2);
  __bf16* hbuf  = (__bf16*)(ws + o); o = align256(o + (size_t)E_ * CAP_ * F_ * 2);
  float*  obuf  = (float*) (ws + o); o = align256(o + (size_t)E_ * CAP_ * H_ * 4);
  __bf16* shbuf = (__bf16*)(ws + o); o = align256(o + (size_t)NTOK * FS_ * 2);
  float*  shout = (float*) (ws + o); o = align256(o + (size_t)NTOK * H_ * 4);

  // 1) x -> bf16 (for shared expert GEMMs)
  k_cvt_bf16<<<(NTOK * H_) / (256 * 4), 256, 0, stream>>>(x, xbf);
  // 2) gate / top-k
  k_gate<<<NTOK / 4, 128, 0, stream>>>(x, gate_w, gate_bias, tidx, tw);
  // 3) dispatch
  k_reset<<<1, 32, 0, stream>>>(cnt);
  k_scatter<<<(NTOK * K_) / 8, 256, 0, stream>>>(x, tidx, cnt, bufx, slots);
  // 4) routed experts: h = silu(buf@w1)*(buf@w3); ob = h@w2
  k_gemm_swiglu<<<dim3(F_ / 64, CAP_ / 128, E_), 256, 0, stream>>>(
      bufx, w1, w3, hbuf, H_, F_,
      (long)CAP_ * H_, (long)H_ * F_, (long)CAP_ * F_);
  k_gemm_plain<<<dim3(H_ / 64, CAP_ / 128, E_), 256, 0, stream>>>(
      hbuf, w2, obuf, F_, H_,
      (long)CAP_ * F_, (long)F_ * H_, (long)CAP_ * H_);
  // 5) shared expert: sh = silu(x@Wg)*(x@Wu); shout = sh@Wd
  k_gemm_swiglu<<<dim3(FS_ / 64, NTOK / 128, 1), 256, 0, stream>>>(
      xbf, sh_gate, sh_up, shbuf, H_, FS_, 0, 0, 0);
  k_gemm_plain<<<dim3(H_ / 64, NTOK / 128, 1), 256, 0, stream>>>(
      shbuf, sh_down, shout, FS_, H_, 0, 0, 0);
  // 6) combine
  k_combine<<<NTOK, 256, 0, stream>>>(shout, obuf, tidx, tw, slots, out);
}
